// PixelSelfAttention_78804059947466
// MI455X (gfx1250) — matrix operand order repaired and checked by reading, not compiled, not run on hardware
//
#include <hip/hip_runtime.h>

// ---------------- problem constants ----------------
constexpr int DIMC = 256;          // channels
constexpr int HWSZ = 65536;        // H*W
constexpr int TOK  = 128;          // tokens per block (32 windows * 4)
constexpr float EPSV = 1e-5f;
constexpr float SCALE = 0.35355339059327373f; // 1/sqrt(8)

// ---------------- LDS layout ----------------
constexpr int STRX = 264;  // bf16 stride: xs / ys rows (256 + 8 pad)
constexpr int STRQ = 200;  // bf16 stride: qkv rows (192 + 8 pad)
constexpr int STRA = 72;   // bf16 stride: attn-out rows (64 + 8 pad)
constexpr int STRT = 68;   // f32 stride: transpose buffer (64 + 4 pad)

constexpr int OFF_XS   = 0;
constexpr int OFF_QKV  = OFF_XS  + TOK * STRX * 2;   // 67584
constexpr int OFF_ATT  = OFF_QKV + TOK * STRQ * 2;   // 118784
constexpr int OFF_PSUM = OFF_ATT + TOK * STRA * 2;   // 137216
constexpr int OFF_PSQ  = OFF_PSUM + 256 * 4;         // 138240
constexpr int OFF_MRS  = OFF_PSQ  + 256 * 4;         // 139264
constexpr int OFF_TR   = OFF_QKV;                    // trbuf aliases dead qkv region
constexpr int SMEM_BYTES = OFF_MRS + 256 * 4;        // 140288
static_assert(TOK * STRT * 4 <= TOK * STRQ * 2, "trbuf must fit in qkv region");

// ---------------- types ----------------
typedef __attribute__((ext_vector_type(16))) __bf16 v16bf;
typedef __attribute__((ext_vector_type(8)))  float  v8f;
typedef __attribute__((ext_vector_type(4)))  unsigned int u32x4;
typedef __attribute__((ext_vector_type(4)))  float  f32x4;

union Frag { v16bf v; u32x4 q[2]; };

__device__ __forceinline__ v8f wmma_bf16(v16bf a, v16bf b, v8f c) {
    return __builtin_amdgcn_wmma_f32_16x16x32_bf16(false, a, false, b, (short)0, c, false, false);
}

// ---------------- weight bf16 pre-pack ----------------
__global__ void prep_weights(const float* __restrict__ wq, const float* __restrict__ wo,
                             const float* __restrict__ wc,
                             __bf16* __restrict__ oq, __bf16* __restrict__ oo,
                             __bf16* __restrict__ oc) {
    int i = blockIdx.x * 256 + threadIdx.x;      // 65536 threads total
    if (i < 192 * 256) oq[i] = (__bf16)wq[i];
    if (i < 64  * 256) oo[i] = (__bf16)wo[i];
    if (i < 256 * 256) oc[i] = (__bf16)wc[i];
}

// ---------------- fused LN + QKV + attention + out-proj + 1x1 conv ----------------
__global__ __launch_bounds__(256)
void fused_pixel_attn(const float* __restrict__ x,
                      const float* __restrict__ pos,
                      const float* __restrict__ lnw,
                      const float* __restrict__ lnb,
                      const float* __restrict__ bqkv,
                      const float* __restrict__ bout,
                      const float* __restrict__ bconv,
                      const __bf16* __restrict__ wq_bf,
                      const __bf16* __restrict__ wo_bf,
                      const __bf16* __restrict__ wc_bf,
                      float* __restrict__ out) {
    extern __shared__ char smem[];
    __bf16* xs    = (__bf16*)(smem + OFF_XS);    // [128][STRX]  input -> later y (conv input)
    __bf16* qkvs  = (__bf16*)(smem + OFF_QKV);   // [128][STRQ]
    __bf16* attos = (__bf16*)(smem + OFF_ATT);   // [128][STRA]
    float*  psum  = (float*) (smem + OFF_PSUM);  // [256]
    float*  psq   = (float*) (smem + OFF_PSQ);   // [256]
    float*  mrs   = (float*) (smem + OFF_MRS);   // [128][2] mean,rstd per token
    float*  trbuf = (float*) (smem + OFF_TR);    // [128][STRT] f32, aliases qkvs

    const int tid = threadIdx.x;
    const int c0 = blockIdx.x;    // 0..3   (column block of 32 windows)
    const int r  = blockIdx.y;    // 0..127 (window row)
    const int b  = blockIdx.z;    // 0..3

    // ---- Phase 1: gather x (channel-major) -> token-major LDS, +pos, LN stats ----
    const int pix   = tid & 127;            // 0..127 pixels of this 2x64 strip
    const int dpar  = tid >> 7;             // d parity handled by this thread
    const int row01 = pix >> 6;             // which of the 2 rows
    const int col   = pix & 63;             // column within strip
    const int tl    = row01 * 2 + (col & 1);            // token index in window
    const int token = (col >> 1) * 4 + tl;              // 0..127 local token id
    const int hrow  = r * 2 + row01;
    const int wcol  = c0 * 64 + col;
    const float* xrow = x + (size_t)b * DIMC * HWSZ + (size_t)hrow * 256 + wcol;

    float s1 = 0.f, s2 = 0.f;
    for (int i = 0; i < 128; ++i) {
        int d = i * 2 + dpar;
        float v = xrow[(size_t)d * HWSZ] + pos[tl * DIMC + d];
        __bf16 bv = (__bf16)v;
        xs[token * STRX + d] = bv;
        float fv = (float)bv;
        s1 += fv; s2 += fv * fv;
    }
    psum[tid] = s1; psq[tid] = s2;
    __syncthreads();
    if (tid < 128) {
        float sum = psum[tid] + psum[tid + 128];
        float sq  = psq[tid]  + psq[tid + 128];
        float mu  = sum * (1.f / 256.f);
        float var = sq * (1.f / 256.f) - mu * mu;
        float rstd = rsqrtf(var + EPSV);
        int prow = tid >> 6, pcol = tid & 63;
        int tk = (pcol >> 1) * 4 + prow * 2 + (pcol & 1);
        mrs[tk * 2] = mu; mrs[tk * 2 + 1] = rstd;
    }
    __syncthreads();
    {   // normalize in place
        float mu = mrs[token * 2], rstd = mrs[token * 2 + 1];
        for (int i = 0; i < 128; ++i) {
            int d = i * 2 + dpar;
            float v = (float)xs[token * STRX + d];
            v = (v - mu) * rstd * lnw[d] + lnb[d];
            xs[token * STRX + d] = (__bf16)v;
        }
    }
    __syncthreads();

    // ---- wave-level WMMA coordinates ----
    const int lane = tid & 31;
    const int wave = tid >> 5;            // 0..7, each owns 16 tokens
    const int mrow = lane & 15;
    const int hiA  = lane >> 4;           // 0/1
    const int tokA = wave * 16 + mrow;    // A-fragment row (token)

    // ---- Phase 2: QKV GEMM  (M=128, K=256, N=192), bf16 WMMA ----
    {
        v8f acc[12];
        #pragma unroll
        for (int nt = 0; nt < 12; ++nt) acc[nt] = v8f{0.f,0.f,0.f,0.f,0.f,0.f,0.f,0.f};
        for (int kt = 0; kt < 8; ++kt) {
            Frag a;
            const __bf16* ap = xs + tokA * STRX + kt * 32 + hiA * 8;
            a.q[0] = *(const u32x4*)(ap);
            a.q[1] = *(const u32x4*)(ap + 16);
            #pragma unroll
            for (int nt = 0; nt < 12; ++nt) {
                int n = nt * 16 + mrow;
                Frag bb;
                const __bf16* bp = wq_bf + n * 256 + kt * 32 + hiA * 8;
                bb.q[0] = *(const u32x4*)(bp);
                bb.q[1] = *(const u32x4*)(bp + 16);
                acc[nt] = wmma_bf16(a.v, bb.v, acc[nt]);
            }
        }
        #pragma unroll
        for (int nt = 0; nt < 12; ++nt) {
            int n = nt * 16 + mrow;
            float bias = bqkv[n];
            #pragma unroll
            for (int rr = 0; rr < 8; ++rr) {
                int m = rr + hiA * 8;
                qkvs[(wave * 16 + m) * STRQ + n] = (__bf16)(acc[nt][rr] + bias);
            }
        }
    }
    __syncthreads();

    // ---- Phase 3: tiny SDPA, 1 thread = 1 (window, head) ----
    {
        int win = tid >> 3, head = tid & 7;
        int base = (win * 4) * STRQ + head * 8;
        float qv[4][8], kv[4][8], vv[4][8], ov[4][8];
        #pragma unroll
        for (int t = 0; t < 4; ++t)
            #pragma unroll
            for (int j = 0; j < 8; ++j) {
                qv[t][j] = (float)qkvs[base + t * STRQ + j];
                kv[t][j] = (float)qkvs[base + t * STRQ + 64 + j];
                vv[t][j] = (float)qkvs[base + t * STRQ + 128 + j];
            }
        #pragma unroll
        for (int t = 0; t < 4; ++t) {
            float sc[4], mx = -1e30f;
            #pragma unroll
            for (int s = 0; s < 4; ++s) {
                float d = 0.f;
                #pragma unroll
                for (int j = 0; j < 8; ++j) d += qv[t][j] * kv[s][j];
                sc[s] = d * SCALE; mx = fmaxf(mx, sc[s]);
            }
            float den = 0.f;
            #pragma unroll
            for (int s = 0; s < 4; ++s) { sc[s] = __expf(sc[s] - mx); den += sc[s]; }
            float rden = 1.f / den;
            #pragma unroll
            for (int j = 0; j < 8; ++j) {
                float a = 0.f;
                #pragma unroll
                for (int s = 0; s < 4; ++s) a += sc[s] * vv[s][j];
                ov[t][j] = a * rden;
            }
        }
        #pragma unroll
        for (int t = 0; t < 4; ++t)
            #pragma unroll
            for (int j = 0; j < 8; ++j)
                attos[(win * 4 + t) * STRA + head * 8 + j] = (__bf16)ov[t][j];
    }
    __syncthreads();

    // ---- Phase 4: out-proj GEMM (M=128, K=64, N=256) -> y into xs (reuse) ----
    {
        v8f acc[16];
        #pragma unroll
        for (int nt = 0; nt < 16; ++nt) acc[nt] = v8f{0.f,0.f,0.f,0.f,0.f,0.f,0.f,0.f};
        #pragma unroll
        for (int kt = 0; kt < 2; ++kt) {
            Frag a;
            const __bf16* ap = attos + tokA * STRA + kt * 32 + hiA * 8;
            a.q[0] = *(const u32x4*)(ap);
            a.q[1] = *(const u32x4*)(ap + 16);
            #pragma unroll
            for (int nt = 0; nt < 16; ++nt) {
                int n = nt * 16 + mrow;
                Frag bb;
                const __bf16* bp = wo_bf + n * 64 + kt * 32 + hiA * 8;
                bb.q[0] = *(const u32x4*)(bp);
                bb.q[1] = *(const u32x4*)(bp + 16);
                acc[nt] = wmma_bf16(a.v, bb.v, acc[nt]);
            }
        }
        // write y (bf16) into xs rows owned by this wave (same-wave RAW only)
        #pragma unroll
        for (int nt = 0; nt < 16; ++nt) {
            int n = nt * 16 + mrow;
            float bias = bout[n];
            #pragma unroll
            for (int rr = 0; rr < 8; ++rr) {
                int m = rr + hiA * 8;
                xs[(wave * 16 + m) * STRX + n] = (__bf16)(acc[nt][rr] + bias);
            }
        }
    }
    __syncthreads();

    // ---- Phase 5: 1x1 conv GEMM (M=128, K=256, N=256) ----
    v8f acc[16];
    #pragma unroll
    for (int nt = 0; nt < 16; ++nt) acc[nt] = v8f{0.f,0.f,0.f,0.f,0.f,0.f,0.f,0.f};
    for (int kt = 0; kt < 8; ++kt) {
        Frag a;
        const __bf16* ap = xs + tokA * STRX + kt * 32 + hiA * 8;
        a.q[0] = *(const u32x4*)(ap);
        a.q[1] = *(const u32x4*)(ap + 16);
        #pragma unroll
        for (int nt = 0; nt < 16; ++nt) {
            int n = nt * 16 + mrow;
            Frag bb;
            const __bf16* bp = wc_bf + n * 256 + kt * 32 + hiA * 8;
            bb.q[0] = *(const u32x4*)(bp);
            bb.q[1] = *(const u32x4*)(bp + 16);
            acc[nt] = wmma_bf16(a.v, bb.v, acc[nt]);
        }
    }

    // ---- Phase 6: transpose through LDS, coalesced store to (B, C, H, W) ----
    #pragma unroll 1
    for (int c = 0; c < 4; ++c) {               // 4 channel chunks of 64
        __syncthreads();                        // previous chunk reads done
        #pragma unroll
        for (int q4 = 0; q4 < 4; ++q4) {
            int nt = c * 4 + q4;
            int loc = q4 * 16 + mrow;           // channel within chunk
            float bias = bconv[c * 64 + loc];
            #pragma unroll
            for (int rr = 0; rr < 8; ++rr) {
                int m = rr + hiA * 8;
                trbuf[(wave * 16 + m) * STRT + loc] = acc[nt][rr] + bias;
            }
        }
        __syncthreads();
        #pragma unroll
        for (int i = 0; i < 8; ++i) {           // 2048 float4 stores / 256 threads
            int idx  = tid + 256 * i;
            int oloc = idx >> 5;
            int rest = idx & 31;
            int row  = rest >> 4;
            int g    = rest & 15;               // 4-pixel group along the row
            int t0   = 8 * g + 2 * row;
            f32x4 v;
            v[0] = trbuf[(t0    ) * STRT + oloc];
            v[1] = trbuf[(t0 + 1) * STRT + oloc];
            v[2] = trbuf[(t0 + 4) * STRT + oloc];
            v[3] = trbuf[(t0 + 5) * STRT + oloc];
            int o = c * 64 + oloc;
            size_t addr = ((size_t)b * DIMC + o) * HWSZ
                        + (size_t)(r * 2 + row) * 256 + (c0 * 64 + 4 * g);
            *(f32x4*)(out + addr) = v;
        }
    }
}

// ---------------- launch ----------------
extern "C" void kernel_launch(void* const* d_in, const int* in_sizes, int n_in,
                              void* d_out, int out_size, void* d_ws, size_t ws_size,
                              hipStream_t stream) {
    const float* x     = (const float*)d_in[0];
    const float* pos   = (const float*)d_in[1];
    const float* lnw   = (const float*)d_in[2];
    const float* lnb   = (const float*)d_in[3];
    const float* wqkv  = (const float*)d_in[4];
    const float* bqkv  = (const float*)d_in[5];
    const float* wout  = (const float*)d_in[6];
    const float* bout  = (const float*)d_in[7];
    const float* wconv = (const float*)d_in[8];
    const float* bconv = (const float*)d_in[9];

    char* ws = (char*)d_ws;
    __bf16* wq_bf = (__bf16*)(ws + 0);        // 192*256*2 = 98304
    __bf16* wo_bf = (__bf16*)(ws + 98304);    // 256*64*2  = 32768
    __bf16* wc_bf = (__bf16*)(ws + 131072);   // 256*256*2 = 131072

    prep_weights<<<256, 256, 0, stream>>>(wqkv, wout, wconv, wq_bf, wo_bf, wc_bf);

    (void)hipFuncSetAttribute((const void*)fused_pixel_attn,
                              hipFuncAttributeMaxDynamicSharedMemorySize, SMEM_BYTES);
    dim3 grid(4, 128, 4);   // (col-block, window-row, batch)
    fused_pixel_attn<<<grid, 256, SMEM_BYTES, stream>>>(
        x, pos, lnw, lnb, bqkv, bout, bconv, wq_bf, wo_bf, wc_bf, (float*)d_out);
}
